// Gate_86689619902654
// MI455X (gfx1250) — matrix operand order repaired and checked by reading, not compiled, not run on hardware
//
#include <hip/hip_runtime.h>
#include <hip/hip_bf16.h>

typedef __attribute__((ext_vector_type(16))) _Float16 v16h;
typedef __attribute__((ext_vector_type(8)))  _Float16 v8h;
typedef __attribute__((ext_vector_type(8)))  float    v8f;
typedef __attribute__((ext_vector_type(4)))  float    f4;

// Workspace layout (bytes):
//   [0,        2 MiB)   pooled_h : f16 [256][4096]  (gate_input)
//   [+2 MiB,  +256KiB)  w1t      : f16 [32][4096]   (W1 transposed)
//   [..,      +16 KiB)  h_h      : f16 [256][32]    (hidden, post-ReLU)
//   [..,      +64 KiB)  logits   : f32 [256][64]
#define POOLED_OFF  0u
#define W1T_OFF     (2u * 1024u * 1024u)
#define HH_OFF      (W1T_OFF + 256u * 1024u)
#define LOGITS_OFF  (HH_OFF + 16u * 1024u)

// ---------------------------------------------------------------------------
// K1: 4x4 adaptive average pool. One block per (b,c) 64x64 image.
// Thread t = r*4+q sums 16 contiguous floats (row r, quarter q) via b128
// loads; LDS reduce over 16 rows gives each of the 16 block means.
// ---------------------------------------------------------------------------
__global__ void pool_kernel(const float* __restrict__ x,
                            _Float16* __restrict__ pooled) {
  __shared__ float part[256];
  const int img = blockIdx.x;            // b*256 + c
  const int t   = threadIdx.x;
  const int r   = t >> 2;
  const int q   = t & 3;
  const float* p = x + (size_t)img * 4096 + r * 64 + q * 16;
  f4 a0 = *(const f4*)(p);
  f4 a1 = *(const f4*)(p + 4);
  f4 a2 = *(const f4*)(p + 8);
  f4 a3 = *(const f4*)(p + 12);
  part[t] = (a0.x + a0.y + a0.z + a0.w) + (a1.x + a1.y + a1.z + a1.w) +
            (a2.x + a2.y + a2.z + a2.w) + (a3.x + a3.y + a3.z + a3.w);
  __syncthreads();
  if (t < 16) {
    const int bi = t >> 2, qq = t & 3;
    float acc = 0.0f;
#pragma unroll
    for (int rr = 0; rr < 16; ++rr) acc += part[(bi * 16 + rr) * 4 + qq];
    const int b = img >> 8, c = img & 255;
    // gate_input index: b*4096 + c*16 + i*4 + j
    pooled[(size_t)b * 4096 + c * 16 + bi * 4 + qq] =
        (_Float16)(acc * (1.0f / 256.0f));
  }
}

// ---------------------------------------------------------------------------
// K2: W1 [4096,32] f32 -> w1t [32,4096] f16 (so WMMA B-fragments are
// contiguous 128-bit loads).
// ---------------------------------------------------------------------------
__global__ void w1t_kernel(const float* __restrict__ W1,
                           _Float16* __restrict__ w1t) {
  const int idx = blockIdx.x * 256 + threadIdx.x;  // 0..131071
  const int k = idx >> 5, n = idx & 31;
  w1t[(size_t)n * 4096 + k] = (_Float16)W1[(size_t)k * 32 + n];
}

// ---------------------------------------------------------------------------
// K3: GEMM1: pooled[256,4096] x W1[4096,32] + b1, ReLU -> h_h f16 [256,32].
// 32 waves, each owns one 16x16 output tile; K-loop of 128 WMMAs.
// ---------------------------------------------------------------------------
__global__ void gemm1_kernel(const _Float16* __restrict__ pooled,
                             const _Float16* __restrict__ w1t,
                             const float* __restrict__ b1,
                             _Float16* __restrict__ hout) {
  const int wave = (blockIdx.x * blockDim.x + threadIdx.x) >> 5;  // 0..31
  const int lane = threadIdx.x & 31;
  const int mt = wave >> 1;            // 16 M tiles
  const int nt = wave & 1;             // 2 N tiles
  const int lm = lane & 15;
  const int hi = (lane >= 16) ? 1 : 0;
  const int row   = mt * 16 + lm;      // A row this lane supplies
  const int ncol  = nt * 16 + lm;      // B/D column this lane supplies
  const int abase = hi ? 8 : 0;        // A per-lane K base (16-bit A layout)
  const int bbase = hi ? 8 : 0;        // B per-lane K base
  const _Float16* Arow = pooled + (size_t)row * 4096;
  const _Float16* Brow = w1t + (size_t)ncol * 4096 + bbase;
  v8f acc = {};
  for (int k0 = 0; k0 < 4096; k0 += 32) {
    // A elements 0..7 -> K = abase..abase+7 ; 8..15 -> K = abase+16..abase+23
    v8h alo = *(const v8h*)(Arow + k0 + abase);
    v8h ahi = *(const v8h*)(Arow + k0 + abase + 16);
    v16h a;
#pragma unroll
    for (int i = 0; i < 8; ++i) { a[i] = alo[i]; a[i + 8] = ahi[i]; }
    // B elements 0..7 -> K = bbase..bbase+7 ; 8..15 -> K = bbase+16..bbase+23
    v8h blo = *(const v8h*)(Brow + k0);
    v8h bhi = *(const v8h*)(Brow + k0 + 16);
    v16h bfrag;
#pragma unroll
    for (int i = 0; i < 8; ++i) { bfrag[i] = blo[i]; bfrag[i + 8] = bhi[i]; }
    acc = __builtin_amdgcn_wmma_f32_16x16x32_f16(
        false, a, false, bfrag, (short)0, acc, false, false);
  }
  const float bias = b1[ncol];
#pragma unroll
  for (int v = 0; v < 8; ++v) {
    const int m = mt * 16 + v + (hi ? 8 : 0);   // D: M = vgpr + half*8
    float val = acc[v] + bias;
    val = val > 0.0f ? val : 0.0f;              // ReLU
    hout[(size_t)m * 32 + ncol] = (_Float16)val;
  }
}

// ---------------------------------------------------------------------------
// K4: GEMM2: h[256,32] x W2[32,64] + b2 -> logits f32 [256,64].
// 64 waves, one WMMA each (K==32).
// ---------------------------------------------------------------------------
__global__ void gemm2_kernel(const _Float16* __restrict__ hh,
                             const float* __restrict__ W2,
                             const float* __restrict__ b2,
                             float* __restrict__ logits) {
  const int wave = (blockIdx.x * blockDim.x + threadIdx.x) >> 5;  // 0..63
  const int lane = threadIdx.x & 31;
  const int mt = wave >> 2;            // 16 M tiles
  const int nt = wave & 3;             // 4 N tiles
  const int lm = lane & 15;
  const int hi = (lane >= 16) ? 1 : 0;
  const int row   = mt * 16 + lm;
  const int ncol  = nt * 16 + lm;
  const int abase = hi ? 8 : 0;
  const int bbase = hi ? 8 : 0;
  const _Float16* Arow = hh + (size_t)row * 32;
  v8h alo = *(const v8h*)(Arow + abase);
  v8h ahi = *(const v8h*)(Arow + abase + 16);
  v16h a;
#pragma unroll
  for (int i = 0; i < 8; ++i) { a[i] = alo[i]; a[i + 8] = ahi[i]; }
  v16h bfrag;
#pragma unroll
  for (int j = 0; j < 16; ++j)
    bfrag[j] = (_Float16)W2[(size_t)(bbase + j + (j >= 8 ? 8 : 0)) * 64 + ncol];
  v8f acc = {};
  acc = __builtin_amdgcn_wmma_f32_16x16x32_f16(
      false, a, false, bfrag, (short)0, acc, false, false);
  const float bias = b2[ncol];
#pragma unroll
  for (int v = 0; v < 8; ++v) {
    const int m = mt * 16 + v + (hi ? 8 : 0);
    logits[(size_t)m * 64 + ncol] = acc[v] + bias;
  }
}

// ---------------------------------------------------------------------------
// K5: per-row top-2, softmax over the pair, scatter into gates; indices
// appended (as float) after the 16384 gate elements.
// ---------------------------------------------------------------------------
__global__ void topk_kernel(const float* __restrict__ logits,
                            float* __restrict__ out) {
  const int b = blockIdx.x * blockDim.x + threadIdx.x;
  if (b >= 256) return;
  const float* lr = logits + (size_t)b * 64;
  float m1 = -3.4e38f; int i1 = 0;
  for (int e = 0; e < 64; ++e) {
    const float v = lr[e];
    if (v > m1) { m1 = v; i1 = e; }        // first-occurrence max
  }
  float m2 = -3.4e38f; int i2 = 0;
  for (int e = 0; e < 64; ++e) {
    if (e == i1) continue;
    const float v = lr[e];
    if (v > m2) { m2 = v; i2 = e; }
  }
  // softmax([m1, m2]) with m1 >= m2 (stable)
  const float p0 = 1.0f / (1.0f + __expf(m2 - m1));
  const float p1 = 1.0f - p0;
  float* g = out + (size_t)b * 64;
  for (int e = 0; e < 64; ++e) g[e] = 0.0f;
  g[i1] = p0;
  g[i2] = p1;
  float* idx = out + 256 * 64;
  idx[b * 2 + 0] = (float)i1;
  idx[b * 2 + 1] = (float)i2;
}

// ---------------------------------------------------------------------------
extern "C" void kernel_launch(void* const* d_in, const int* in_sizes, int n_in,
                              void* d_out, int out_size, void* d_ws, size_t ws_size,
                              hipStream_t stream) {
  const float* x  = (const float*)d_in[0];   // [256,256,64,64]
  const float* W1 = (const float*)d_in[1];   // [4096,32]
  const float* b1 = (const float*)d_in[2];   // [32]
  const float* W2 = (const float*)d_in[3];   // [32,64]
  const float* b2 = (const float*)d_in[4];   // [64]
  float* out = (float*)d_out;                // 16384 gates + 512 indices

  char* ws = (char*)d_ws;
  _Float16* pooled = (_Float16*)(ws + POOLED_OFF);
  _Float16* w1t    = (_Float16*)(ws + W1T_OFF);
  _Float16* hh     = (_Float16*)(ws + HH_OFF);
  float*    logits = (float*)(ws + LOGITS_OFF);

  pool_kernel <<<dim3(256 * 256), dim3(256), 0, stream>>>(x, pooled);
  w1t_kernel  <<<dim3(512),       dim3(256), 0, stream>>>(W1, w1t);
  gemm1_kernel<<<dim3(4),         dim3(256), 0, stream>>>(pooled, w1t, b1, hh);
  gemm2_kernel<<<dim3(8),         dim3(256), 0, stream>>>(hh, W2, b2, logits);
  topk_kernel <<<dim3(1),         dim3(256), 0, stream>>>(logits, out);
}
